// PSAEBFrameDenoiser_20117626814825
// MI455X (gfx1250) — matrix operand order repaired and checked by reading, not compiled, not run on hardware
//
#include <hip/hip_runtime.h>
#include <stdint.h>
#include <stddef.h>

// ---------------- CDNA5 WMMA types ----------------
typedef __attribute__((ext_vector_type(16))) __bf16 v16bf;
typedef __attribute__((ext_vector_type(8)))  float  v8f;

// ---------------- model dims ----------------
#define MD_N   2048
#define MD_E   102400
#define MD_E2  61440
#define MD_WL  0.5773502691896258f   // sqrt(1/3)
#define MD_WC  0.16666666666666666f  // sqrt(2/(9*8)) = 1/6
#define LN1E4  9.210340371976184f    // ln(10000)

static __device__ __forceinline__ __bf16 f2bf(float f) {
  unsigned u = __builtin_bit_cast(unsigned, f);
  unsigned r = (u + 0x7FFFu + ((u >> 16) & 1u)) >> 16;
  unsigned short s = (unsigned short)r;
  return __builtin_bit_cast(__bf16, s);
}
static __device__ __forceinline__ float bf2f(__bf16 b) {
  unsigned short s = __builtin_bit_cast(unsigned short, b);
  unsigned u = ((unsigned)s) << 16;
  return __builtin_bit_cast(float, u);
}
// order-preserving float<->uint for atomicMax-based segment_max
static __device__ __forceinline__ unsigned fenc(float f) {
  int i = __float_as_int(f);
  return (i >= 0) ? ((unsigned)i | 0x80000000u) : ~((unsigned)i);
}
static __device__ __forceinline__ float fdec(unsigned u) {
  int i = (u & 0x80000000u) ? (int)(u & 0x7FFFFFFFu) : ~((int)u);
  return __int_as_float(i);
}

// ---------------- WMMA GEMM: D[M,Nc] = A[M,K](bf16) @ B[K,Np](bf16) ----------------
// Preconditions (arranged by host): M % 64 == 0; K % 32 == 0 (A zero-padded);
// B stored K x Np with Np = ceil64(Nc), zero-padded. All loads are guard-free.
// 256 threads = 8 waves (4x2); block tile 64x64; each wave 16x32 -> 2 WMMA/iter.
// A tile staged via GLOBAL_LOAD_ASYNC_TO_LDS_B128 (ASYNCcnt), double-buffered LDS:
// tile t+1 is in flight while tile t feeds the WMMAs; s_wait_asynccnt 1 drains only
// the older transfer (async loads complete in order per CDNA5 ISA).
__global__ __launch_bounds__(256)
void k_gemm(const __bf16* __restrict__ A, const __bf16* __restrict__ B,
            float* __restrict__ Df, __bf16* __restrict__ Dbf,
            int M, int Nc, int K, int act)
{
  const int Np = (Nc + 63) & ~63;
  __shared__ __align__(16) __bf16 As[2][64][40];   // row-major, stride 40 (80B, 16B aligned)
  __shared__ __align__(16) __bf16 Bs[2][64][40];   // transposed: [n][k]
  const int tid  = threadIdx.x;
  const int lane = tid & 31, wave = tid >> 5;
  const int wm = wave >> 1, wn = wave & 1;      // 4x2 wave grid
  const int m0 = blockIdx.x * 64, n0 = blockIdx.y * 64;
  const int hsel = lane >> 4, l16 = lane & 15;
  const int arow = tid >> 2, acol = (tid & 3) * 8;   // A: 64 rows x 32 k, 8 bf16/thread
  const int bkr  = tid >> 3, bnc  = (tid & 7) * 8;   // B: 32 k x 64 n, 8 bf16/thread
  const __bf16* aptr = A + (size_t)(m0 + arow) * K + acol;
  const __bf16* bptr = B + (size_t)bkr * Np + (n0 + bnc);
  v8f acc0 = {}, acc1 = {};

  const unsigned ldsA0 = (unsigned)(size_t)&As[0][arow][acol];
  const unsigned ldsA1 = (unsigned)(size_t)&As[1][arow][acol];
  auto issueA = [&](int k0, int buf) {
    unsigned lds = buf ? ldsA1 : ldsA0;
    unsigned long long ga = (unsigned long long)(size_t)(aptr + k0);
    asm volatile("global_load_async_to_lds_b128 %0, %1, off"
                 :: "v"(lds), "v"(ga) : "memory");
  };
  auto stageB = [&](int k0, int buf) {
    uint4 bv = *(const uint4*)(bptr + (size_t)k0 * Np);
    const __bf16* bsrc = (const __bf16*)&bv;
#pragma unroll
    for (int j = 0; j < 8; ++j) Bs[buf][bnc + j][bkr] = bsrc[j];
  };

  issueA(0, 0);
  stageB(0, 0);
  int cur = 0;
  for (int k0 = 0; k0 < K; k0 += 32) {
    const int nxt = cur ^ 1;
    if (k0 + 32 < K) {
      issueA(k0 + 32, nxt);                              // overlap next A tile
      stageB(k0 + 32, nxt);                              // overlap next B tile
      asm volatile("s_wait_asynccnt 0x1" ::: "memory");  // current A tile landed
    } else {
      asm volatile("s_wait_asynccnt 0x0" ::: "memory");
    }
    __syncthreads();
    // A fragment: row = wm*16+l16; elems = K(hsel*16 .. hsel*16+15) contiguous
    v16bf af, bf0, bf1;
    const __bf16* ar_ = &As[cur][wm * 16 + l16][0];
    ((uint4*)&af)[0] = *(const uint4*)(ar_ + hsel * 16);
    ((uint4*)&af)[1] = *(const uint4*)(ar_ + hsel * 16 + 8);
    // B fragments: col = wn*32 + sub*16 + l16; elems = K(hsel*16 .. hsel*16+15) contiguous
    const __bf16* b0 = &Bs[cur][wn * 32 + l16][hsel * 16];
    const __bf16* b1 = &Bs[cur][wn * 32 + 16 + l16][hsel * 16];
    ((uint4*)&bf0)[0] = *(const uint4*)(b0);
    ((uint4*)&bf0)[1] = *(const uint4*)(b0 + 8);
    ((uint4*)&bf1)[0] = *(const uint4*)(b1);
    ((uint4*)&bf1)[1] = *(const uint4*)(b1 + 8);
    acc0 = __builtin_amdgcn_wmma_f32_16x16x32_bf16(false, af, false, bf0, (short)0, acc0, false, false);
    acc1 = __builtin_amdgcn_wmma_f32_16x16x32_bf16(false, af, false, bf1, (short)0, acc1, false, false);
    __syncthreads();
    cur = nxt;
  }
#pragma unroll
  for (int sub = 0; sub < 2; ++sub) {
    int col = n0 + wn * 32 + sub * 16 + l16;
    if (col < Nc) {
#pragma unroll
      for (int r = 0; r < 8; ++r) {
        int row = m0 + wm * 16 + r + hsel * 8;
        float v = sub ? acc1[r] : acc0[r];
        if (act == 1) v = fmaxf(v, 0.f);
        if (Df)  Df [(size_t)row * Nc + col] = v;
        if (Dbf) Dbf[(size_t)row * Nc + col] = f2bf(v);
      }
    }
  }
}

// ---------------- utility kernels ----------------
__global__ void k_cvt(const float* __restrict__ x, __bf16* __restrict__ y, long long n) {
  long long i = (long long)blockIdx.x * 256 + threadIdx.x;
  if (i < n) y[i] = f2bf(x[i]);
}
// weight convert with zero padding: src K x N -> dst Kp x Np
__global__ void k_cvtpad(const float* __restrict__ x, __bf16* __restrict__ y,
                         int K, int N, int Kp, int Np) {
  long long idx = (long long)blockIdx.x * 256 + threadIdx.x;
  if (idx >= (long long)Kp * Np) return;
  int r = (int)(idx / Np), c = (int)(idx % Np);
  float v = (r < K && c < N) ? x[(size_t)r * N + c] : 0.f;
  y[idx] = f2bf(v);
}
__global__ void k_fillf(float* p, float v, long long n) {
  long long i = (long long)blockIdx.x * 256 + threadIdx.x;
  if (i < n) p[i] = v;
}
__global__ void k_fillu(unsigned* p, unsigned v, long long n) {
  long long i = (long long)blockIdx.x * 256 + threadIdx.x;
  if (i < n) p[i] = v;
}

__global__ __launch_bounds__(256) void k_center(const float* rig, float* center, int n) {
  __shared__ float sh[256];
  for (int d = 0; d < 3; ++d) {
    float acc = 0.f;
    for (int i = threadIdx.x; i < n; i += 256) acc += rig[(size_t)i * 7 + 4 + d];
    sh[threadIdx.x] = acc; __syncthreads();
    for (int o = 128; o; o >>= 1) { if (threadIdx.x < o) sh[threadIdx.x] += sh[threadIdx.x + o]; __syncthreads(); }
    if (threadIdx.x == 0) center[d] = sh[0] / (float)n;
    __syncthreads();
  }
}

__global__ void k_initframes(const float* rig, const float* center, const unsigned char* xm,
                             float* q0, float* tr, float* trc, float* keep, int n) {
  int i = blockIdx.x * 256 + threadIdx.x;
  if (i >= n) return;
  float qw = rig[i*7], qx = rig[i*7+1], qy = rig[i*7+2], qz = rig[i*7+3];
  float r = rsqrtf(qw*qw + qx*qx + qy*qy + qz*qz);
  q0[i*4] = qw*r; q0[i*4+1] = qx*r; q0[i*4+2] = qy*r; q0[i*4+3] = qz*r;
  for (int d = 0; d < 3; ++d) {
    float c = rig[i*7+4+d] - center[d];
    trc[i*3+d] = c; tr[i*3+d] = c * 0.1f;
  }
  keep[i] = xm[i] ? 0.f : 1.f;
}

__global__ void k_spfeat(const float* X, const int* ei, const int* residx, int E, float* ef) {
  long long idx = (long long)blockIdx.x * 256 + threadIdx.x;
  if (idx >= (long long)E * 128) return;
  int e = (int)(idx >> 7), j = (int)(idx & 127);
  int dst = ei[e], src = ei[E + e];
  float out;
  if (j < 64) {
    float dx = X[dst*3]-X[src*3], dy = X[dst*3+1]-X[src*3+1], dz = X[dst*3+2]-X[src*3+2];
    float dist = sqrtf(dx*dx + dy*dy + dz*dz + 1e-8f);
    float mu = 20.f * (float)j / 63.f;
    float z = (dist - mu) / (20.f / 64.f);
    out = expf(-z*z);
  } else {
    int jj = j - 64;
    float didx = (float)(residx[dst] - residx[src]);
    int i = (jj < 32) ? jj : jj - 32;
    float ang = didx * expf(-LN1E4 * (float)(2*i) / 64.f);
    out = (jj < 32) ? sinf(ang) : cosf(ang);
  }
  ef[idx] = out;
}

// node input, width padded 321 -> 352 (zeros)
__global__ void k_nodeinput(const int* residx, const float* tt, const float* nm, __bf16* out, int n) {
  long long idx = (long long)blockIdx.x * 256 + threadIdx.x;
  if (idx >= (long long)n * 352) return;
  int i = (int)(idx / 352), j = (int)(idx % 352);
  float v = 0.f;
  if (j < 256) {
    int k = (j < 128) ? j : j - 128;
    float ang = (float)residx[i] * expf(-LN1E4 * (float)(2*k) / 256.f);
    v = (j < 128) ? sinf(ang) : cosf(ang);
  } else if (j < 320) {
    int k = j - 256;
    float z = (tt[i] - (float)k / 63.f) / (1.f / 64.f);
    v = expf(-z*z);
  } else if (j == 320) v = nm[i];
  out[idx] = f2bf(v);
}

// out = LN(res + x*keep)*g + b   (res/keep/outbf optional)
__global__ __launch_bounds__(256) void k_ln(const float* __restrict__ x, const float* __restrict__ res,
    const float* __restrict__ keep, const float* __restrict__ g, const float* __restrict__ b,
    float* __restrict__ out, __bf16* __restrict__ outbf, int dim)
{
  __shared__ float sh[256];
  const int row = blockIdx.x, t = threadIdx.x;
  const float kf = keep ? keep[row] : 1.f;
  float v = 0.f;
  if (t < dim) {
    v = x[(size_t)row*dim + t] * kf;
    if (res) v += res[(size_t)row*dim + t];
  }
  sh[t] = v; __syncthreads();
  for (int o = 128; o; o >>= 1) { if (t < o) sh[t] += sh[t+o]; __syncthreads(); }
  const float mean = sh[0] / (float)dim; __syncthreads();
  float d = (t < dim) ? (v - mean) : 0.f;
  sh[t] = d*d; __syncthreads();
  for (int o = 128; o; o >>= 1) { if (t < o) sh[t] += sh[t+o]; __syncthreads(); }
  const float rs = rsqrtf(sh[0] / (float)dim + 1e-5f);
  if (t < dim) {
    float y = d * rs * g[t] + b[t];
    out[(size_t)row*dim + t] = y;
    if (outbf) outbf[(size_t)row*dim + t] = f2bf(y);
  }
}

// nv_out = vln(prev + vu*keep)  (prev optional)
__global__ __launch_bounds__(64) void k_vln(const float* __restrict__ prev, const float* __restrict__ vu,
    const float* __restrict__ keep, const float* __restrict__ g, float* __restrict__ out)
{
  __shared__ float sh[64];
  int i = blockIdx.x, t = threadIdx.x;
  float z = 0.f;
  if (t < 48) {
    float kf = keep ? keep[i] : 1.f;
    z = vu[(size_t)i*48 + t] * kf + (prev ? prev[(size_t)i*48 + t] : 0.f);
  }
  sh[t] = z*z; __syncthreads();
  for (int o = 32; o; o >>= 1) { if (t < o) sh[t] += sh[t+o]; __syncthreads(); }
  float rs = rsqrtf(sh[0] / 16.f + 1e-5f);
  if (t < 48) out[(size_t)i*48 + t] = z * rs * g[t/3];
}

__global__ void k_addscaled(float* dst, const float* x, const float* keep, int dim, long long total) {
  long long idx = (long long)blockIdx.x * 256 + threadIdx.x;
  if (idx >= total) return;
  int row = (int)(idx / dim);
  dst[idx] += x[idx] * (keep ? keep[row] : 1.f);
}
__global__ void k_maskrow(float* dst, __bf16* dbf, const float* keep, int dim, long long total) {
  long long idx = (long long)blockIdx.x * 256 + threadIdx.x;
  if (idx >= total) return;
  int row = (int)(idx / dim);
  float v = dst[idx] * keep[row];
  dst[idx] = v;
  if (dbf) dbf[idx] = f2bf(v);
}

__global__ void k_toglobal(const float* q0, const float* tr, const float* pl, float* pg, int n, int npts) {
  long long idx = (long long)blockIdx.x * 256 + threadIdx.x;
  if (idx >= (long long)n * npts) return;
  int i = (int)(idx / npts);
  const float* q = q0 + (size_t)i*4;
  const float* v = pl + idx*3;
  float qw=q[0], qx=q[1], qy=q[2], qz=q[3];
  float vx=v[0], vy=v[1], vz=v[2];
  float tx = 2.f*(qy*vz - qz*vy), ty = 2.f*(qz*vx - qx*vz), tz = 2.f*(qx*vy - qy*vx);
  pg[idx*3+0] = vx + qw*tx + (qy*tz - qz*ty) + tr[(size_t)i*3+0];
  pg[idx*3+1] = vy + qw*ty + (qz*tx - qx*tz) + tr[(size_t)i*3+1];
  pg[idx*3+2] = vz + qw*tz + (qx*ty - qy*tx) + tr[(size_t)i*3+2];
}

__global__ void k_addvec(float* vpl, const float* nv, const float* wvvec, int n) {
  long long idx = (long long)blockIdx.x * 256 + threadIdx.x;
  if (idx >= (long long)n * 288) return;
  int i = (int)(idx / 288), r = (int)(idx % 288), hp = r / 3, d = r % 3;
  float acc = 0.f;
  const float* np = nv + (size_t)i*48;
  for (int c = 0; c < 16; ++c) acc += np[c*3 + d] * wvvec[c*96 + hp];
  vpl[idx] += acc;
}

// one wave per (edge, head); qk dot + point distance + bias -> logit, atomic segment max
__global__ __launch_bounds__(256) void k_logits(const __bf16* __restrict__ qb, const __bf16* __restrict__ kb,
    const float* __restrict__ qpg, const float* __restrict__ kpg, const float* __restrict__ ebias,
    const float* __restrict__ gamma, const int* __restrict__ ei, int E,
    float* __restrict__ logit, unsigned* __restrict__ segmax)
{
  long long gw = (((long long)blockIdx.x * 256) + threadIdx.x) >> 5;
  int lane = threadIdx.x & 31;
  if (gw >= (long long)E * 8) return;
  int e = (int)(gw >> 3), h = (int)(gw & 7);
  int dst = ei[e], src = ei[E + e];
  const __bf16* qp = qb + (size_t)dst*1024 + h*128;
  const __bf16* kp = kb + (size_t)src*1024 + h*128;
  float qk = 0.f;
#pragma unroll
  for (int j = 0; j < 4; ++j) {
    int c = lane + j*32;
    qk += bf2f(qp[c]) * bf2f(kp[c]);
  }
  float part = qk * 0.08838834764831845f; // 1/sqrt(128)
  float gm = gamma[h];
  float gsp = (gm > 20.f) ? gm : log1pf(expf(gm)); // softplus
  if (lane < 24) {
    const float* a  = qpg + ((size_t)dst*8 + h) * 24;
    const float* bb = kpg + ((size_t)src*8 + h) * 24;
    float dd = a[lane] - bb[lane];
    part -= 0.5f * MD_WC * gsp * dd * dd;
  }
#pragma unroll
  for (int o = 16; o; o >>= 1) part += __shfl_xor(part, o, 32);
  if (lane == 0) {
    float lg = MD_WL * (part + ebias[(size_t)e*8 + h]);
    logit[(size_t)e*8 + h] = lg;
    atomicMax(&segmax[dst*8 + h], fenc(lg));
  }
}

__global__ void k_expsum(const float* logit, const unsigned* segmax, const int* ei, int E,
                         float* araw, float* segsum) {
  long long idx = (long long)blockIdx.x * 256 + threadIdx.x;
  if (idx >= (long long)E * 8) return;
  int e = (int)(idx >> 3), h = (int)(idx & 7);
  int dst = ei[e];
  float a = expf(logit[idx] - fdec(segmax[dst*8 + h]));
  araw[idx] = a;
  atomicAdd(&segsum[dst*8 + h], a);
}

__global__ __launch_bounds__(256) void k_scatter(const float* __restrict__ araw, const float* __restrict__ segsum,
    const int* __restrict__ ei, int E, const float* __restrict__ vf, const float* __restrict__ eff,
    const float* __restrict__ vpg, float* __restrict__ o, float* __restrict__ oe, float* __restrict__ op)
{
  long long gw = (((long long)blockIdx.x * 256) + threadIdx.x) >> 5;
  int lane = threadIdx.x & 31;
  if (gw >= (long long)E * 8) return;
  int e = (int)(gw >> 3), h = (int)(gw & 7);
  int dst = ei[e], src = ei[E + e];
  float alpha = araw[gw] / (segsum[dst*8 + h] + 1e-9f);
  size_t ob = ((size_t)dst*8 + h) * 128;
  size_t vb = ((size_t)src*8 + h) * 128;
  size_t eb = (size_t)e * 128;
#pragma unroll
  for (int j = 0; j < 4; ++j) {
    int c = lane + j*32;
    atomicAdd(&o[ob + c],  alpha * vf[vb + c]);
    atomicAdd(&oe[ob + c], alpha * eff[eb + c]);
  }
  size_t pb = ((size_t)dst*8 + h) * 36;
  size_t ps = ((size_t)src*8 + h) * 36;
  for (int c = lane; c < 36; c += 32)
    atomicAdd(&op[pb + c], alpha * vpg[ps + c]);
}

// build cat = [o | oe | opl | opn] (bf16, 2432) + opl f32; inverse-frame rotation
__global__ __launch_bounds__(256) void k_cat(const float* o, const float* oe, const float* op,
    const float* q0, const float* tr, float* opl, __bf16* cat, int n)
{
  int i = blockIdx.x, t = threadIdx.x;
  size_t cb = (size_t)i * 2432;
  for (int j = t; j < 1024; j += 256) cat[cb + j]        = f2bf(o [(size_t)i*1024 + j]);
  for (int j = t; j < 1024; j += 256) cat[cb + 1024 + j] = f2bf(oe[(size_t)i*1024 + j]);
  if (t < 96) {
    float qw = q0[i*4], qx = -q0[i*4+1], qy = -q0[i*4+2], qz = -q0[i*4+3];
    const float* p = op + (size_t)i*288 + t*3;
    float vx = p[0]-tr[i*3], vy = p[1]-tr[i*3+1], vz = p[2]-tr[i*3+2];
    float tx = 2.f*(qy*vz - qz*vy), ty = 2.f*(qz*vx - qx*vz), tz = 2.f*(qx*vy - qy*vx);
    float rx = vx + qw*tx + (qy*tz - qz*ty);
    float ry = vy + qw*ty + (qz*tx - qx*tz);
    float rz = vz + qw*tz + (qx*ty - qy*tx);
    size_t ob = (size_t)i*288 + t*3;
    opl[ob] = rx; opl[ob+1] = ry; opl[ob+2] = rz;
    cat[cb + 2048 + t*3 + 0] = f2bf(rx);
    cat[cb + 2048 + t*3 + 1] = f2bf(ry);
    cat[cb + 2048 + t*3 + 2] = f2bf(rz);
    cat[cb + 2336 + t] = f2bf(sqrtf(rx*rx + ry*ry + rz*rz + 1e-8f));
  }
}

__global__ void k_vupd(const float* opl, const float* wvout, float* out, int n) {
  long long idx = (long long)blockIdx.x * 256 + threadIdx.x;
  if (idx >= (long long)n * 48) return;
  int i = (int)(idx / 48), r = (int)(idx % 48), c = r / 3, d = r % 3;
  float acc = 0.f;
  const float* p = opl + (size_t)i*288;
  for (int hp = 0; hp < 96; ++hp) acc += p[hp*3 + d] * wvout[hp*16 + c];
  out[idx] = acc;
}

// [s | vec-norms] width padded 272 -> 288 (zeros)
__global__ void k_cat272(const float* s, const float* nv, __bf16* out, int n) {
  long long idx = (long long)blockIdx.x * 256 + threadIdx.x;
  if (idx >= (long long)n * 288) return;
  int i = (int)(idx / 288), j = (int)(idx % 288);
  float v = 0.f;
  if (j < 256) v = s[(size_t)i*256 + j];
  else if (j < 272) {
    const float* p = nv + (size_t)i*48 + (j - 256)*3;
    v = sqrtf(p[0]*p[0] + p[1]*p[1] + p[2]*p[2] + 1e-8f);
  }
  out[idx] = f2bf(v);
}

__global__ void k_gatevupd(const float* graw, const float* nv, const float* wv, float* out, int n) {
  long long idx = (long long)blockIdx.x * 256 + threadIdx.x;
  if (idx >= (long long)n * 48) return;
  int i = (int)(idx / 48), r = (int)(idx % 48), c = r / 3, d = r % 3;
  float g = 1.f / (1.f + expf(-graw[(size_t)i*16 + c]));
  float acc = 0.f;
  const float* np = nv + (size_t)i*48;
  for (int cc = 0; cc < 16; ++cc) acc += np[cc*3 + d] * wv[cc*16 + c];
  out[idx] = g * acc;
}

__global__ void k_ntv(const float* nv, const float* w, float* out, int n) {
  long long idx = (long long)blockIdx.x * 256 + threadIdx.x;
  if (idx >= (long long)n * 48) return;
  int i = (int)(idx / 48), r = (int)(idx % 48), c = r / 3, d = r % 3;
  float acc = nv[idx];
  const float* np = nv + (size_t)i*48;
  for (int cc = 0; cc < 16; ++cc) acc += np[cc*3 + d] * w[cc*16 + c];
  out[idx] = acc;
}

__global__ void k_upd6(const float* s, const float* nv, const float* nm,
                       const float* bbws, const float* bbwv, float* upd, int n) {
  long long idx = (long long)blockIdx.x * 256 + threadIdx.x;
  if (idx >= (long long)n * 6) return;
  int i = (int)(idx / 6), j = (int)(idx % 6);
  float m = nm[i];
  float acc = 0.f;
  const float* sp = s + (size_t)i*256;
  for (int c = 0; c < 256; ++c) acc += sp[c] * bbws[c*6 + j];
  acc *= m;
  int k = j / 3, d = j % 3;
  float acc2 = 0.f;
  const float* np = nv + (size_t)i*48;
  for (int c = 0; c < 16; ++c) acc2 += np[c*3 + d] * bbwv[c*2 + k];
  acc2 *= m;
  upd[idx] = (acc + acc2) * m;
}

__global__ void k_frameupd(float* q0, float* tr, const float* upd, int n) {
  int i = blockIdx.x * 256 + threadIdx.x;
  if (i >= n) return;
  float u0=upd[i*6], u1=upd[i*6+1], u2=upd[i*6+2], u3=upd[i*6+3], u4=upd[i*6+4], u5=upd[i*6+5];
  float dn = rsqrtf(1.f + u0*u0 + u1*u1 + u2*u2);
  float dw = dn, dx = u0*dn, dy = u1*dn, dz = u2*dn;
  float aw=q0[i*4], ax=q0[i*4+1], ay=q0[i*4+2], az=q0[i*4+3];
  float vx=u3, vy=u4, vz=u5;
  float tx = 2.f*(ay*vz - az*vy), ty = 2.f*(az*vx - ax*vz), tz = 2.f*(ax*vy - ay*vx);
  tr[i*3]   += vx + aw*tx + (ay*tz - az*ty);
  tr[i*3+1] += vy + aw*ty + (az*tx - ax*tz);
  tr[i*3+2] += vz + aw*tz + (ax*ty - ay*tx);
  q0[i*4]   = aw*dw - ax*dx - ay*dy - az*dz;
  q0[i*4+1] = aw*dx + ax*dw + ay*dz - az*dy;
  q0[i*4+2] = aw*dy - ax*dz + ay*dw + az*dx;
  q0[i*4+3] = aw*dz + ax*dy - ay*dx + az*dw;
}

__global__ void k_edgecat(const __bf16* ns, const __bf16* efb, const int* ei, int E, __bf16* x) {
  long long idx = (long long)blockIdx.x * 256 + threadIdx.x;
  if (idx >= (long long)E * 384) return;
  int e = (int)(idx / 384), j = (int)(idx % 384);
  __bf16 v;
  if (j < 128)      { int dst = ei[e];     v = ns[(size_t)dst*128 + j]; }
  else if (j < 256) {                      v = efb[(size_t)e*128 + (j - 128)]; }
  else              { int src = ei[E + e]; v = ns[(size_t)src*128 + (j - 256)]; }
  x[idx] = v;
}

__global__ __launch_bounds__(256) void k_out(const float* s, const float* q0, const float* tr,
    const float* center, const float* psi, float* out, int n) {
  int i = blockIdx.x, t = threadIdx.x;
  out[(size_t)i*256 + t] = s[(size_t)i*256 + t];
  size_t rb = (size_t)n * 256;
  if (t < 4)      out[rb + (size_t)i*7 + t] = q0[i*4 + t];
  else if (t < 7) out[rb + (size_t)i*7 + t] = tr[i*3 + (t-4)] * 10.f + center[t-4];
  if (t == 0) {
    float a = psi[i*2], b = psi[i*2+1];
    float r = 1.f / (sqrtf(a*a + b*b) + 1e-8f);
    size_t pb = rb + (size_t)n * 7;
    out[pb + i*2] = a*r; out[pb + i*2 + 1] = b*r;
  }
}

// ---------------- host orchestration ----------------
struct AttnW { const __bf16 *wq,*wk,*wv,*wqp,*wkp,*wvp,*wout,*web; const float *gamma,*wvvec,*wvout; };
struct EtW   { const float *b,*g; const __bf16 *w1,*w2,*wn; };
struct LuW   { const __bf16 *w1,*wg,*ws; const float *wv; };

extern "C" void kernel_launch(void* const* d_in, const int* in_sizes, int n_in,
                              void* d_out_, int out_size, void* d_ws, size_t ws_size,
                              hipStream_t stream)
{
  (void)in_sizes; (void)n_in; (void)out_size; (void)ws_size;
  auto F = [&](int i) -> const float* { return (const float*)d_in[i]; };
  // jax tree-flatten order (sorted dict keys): embed_edge(0..4), embed_node(5..9),
  // layers at 10 + 56*l, tor_w1=234, tor_w2=235, then tensors 236..242
  const float* rig  = F(236);
  const float* tarr = F(237);
  const unsigned char* xm = (const unsigned char*)d_in[238];
  const float* nm = F(239);
  const int* residx = (const int*)d_in[240];
  const int* ei  = (const int*)d_in[241];
  const int* sei = (const int*)d_in[242];
  float* outp = (float*)d_out_;

  const int Nn = MD_N, E = MD_E, E2 = MD_E2;

  uintptr_t wp = (uintptr_t)d_ws;
  auto alloc  = [&](size_t bytes) -> void* { wp = (wp + 255) & ~((uintptr_t)255); void* p = (void*)wp; wp += bytes; return p; };
  auto allocF = [&](size_t n) -> float*   { return (float*)alloc(n * 4); };
  auto allocB = [&](size_t n) -> __bf16*  { return (__bf16*)alloc(n * 2); };
  auto allocU = [&](size_t n) -> unsigned*{ return (unsigned*)alloc(n * 4); };
  auto gb = [](long long n) -> dim3 { return dim3((unsigned)((n + 255) / 256)); };

  // weight convert: K x N -> ceil32(K) x ceil64(N), zero padded
  auto cvtw = [&](const float* src, int K, int N) -> const __bf16* {
    int Kp = (K + 31) & ~31, Np = (N + 63) & ~63;
    __bf16* dst = allocB((size_t)Kp * Np);
    k_cvtpad<<<gb((long long)Kp * Np), 256, 0, stream>>>(src, dst, K, N, Kp, Np);
    return dst;
  };

  // ---- weights -> bf16 (once per launch; deterministic) ----
  AttnW aseq[4], aspw[4]; EtW etw[4], setw[4]; LuW lcuw[4], lruw[4];
  const __bf16 *ntw1[4], *ntw2[4];
  auto mkattn = [&](int a) -> AttnW {
    AttnW w;
    w.gamma = F(a+0);                  w.web  = cvtw(F(a+1), 128, 8);
    w.wk    = cvtw(F(a+2), 256, 1024); w.wkp  = cvtw(F(a+3), 256, 192);
    w.wout  = cvtw(F(a+4), 2432, 256); w.wq   = cvtw(F(a+5), 256, 1024);
    w.wqp   = cvtw(F(a+6), 256, 192);  w.wv   = cvtw(F(a+7), 256, 1024);
    w.wvout = F(a+8);                  w.wvp  = cvtw(F(a+9), 256, 288);
    w.wvvec = F(a+10);
    return w;
  };
  auto mket = [&](int a) -> EtW {
    EtW w; w.b = F(a+0); w.g = F(a+1);
    w.w1 = cvtw(F(a+2), 384, 128); w.w2 = cvtw(F(a+3), 128, 128); w.wn = cvtw(F(a+4), 256, 128);
    return w;
  };
  auto mklu = [&](int a) -> LuW {
    LuW w; w.w1 = cvtw(F(a+0), 272, 128); w.wg = cvtw(F(a+1), 128, 16);
    w.ws = cvtw(F(a+2), 128, 256); w.wv = F(a+3);
    return w;
  };
  for (int l = 0; l < 4; ++l) {
    int lb = 10 + l * 56;
    aseq[l] = mkattn(lb + 0);
    aspw[l] = mkattn(lb + 11);
    etw[l]  = mket(lb + 24);
    lcuw[l] = mklu(lb + 29);
    lruw[l] = mklu(lb + 39);
    ntw1[l] = cvtw(F(lb + 45), 256, 256);
    ntw2[l] = cvtw(F(lb + 46), 256, 256);
    setw[l] = mket(lb + 48);
  }
  const __bf16* eew1 = cvtw(F(2), 128, 128);
  const __bf16* eew2 = cvtw(F(3), 128, 128);
  const __bf16* eew3 = cvtw(F(4), 128, 128);
  const __bf16* enw1 = cvtw(F(7), 321, 256);   // K padded to 352
  const __bf16* enw2 = cvtw(F(8), 256, 256);
  const __bf16* enw3 = cvtw(F(9), 256, 256);
  const __bf16* torw1 = cvtw(F(234), 256, 128);
  const __bf16* torw2 = cvtw(F(235), 128, 2);

  // ---- activations / scratch (all L2-resident) ----
  float* center = allocF(8);
  float* q0  = allocF((size_t)Nn*4);
  float* tr  = allocF((size_t)Nn*3);
  float* trc = allocF((size_t)Nn*3);
  float* keep = allocF(Nn);
  float* s = allocF((size_t)Nn*256);
  __bf16* sbf = allocB((size_t)Nn*256);
  float* nvA = allocF((size_t)Nn*48);
  float* nvB = allocF((size_t)Nn*48);
  float* ef  = allocF((size_t)E*128);   __bf16* efb  = allocB((size_t)E*128);
  float* sef = allocF((size_t)E2*128);  __bf16* sefb = allocB((size_t)E2*128);
  __bf16* qbf = allocB((size_t)Nn*1024);
  __bf16* kbf = allocB((size_t)Nn*1024);
  float* vf  = allocF((size_t)Nn*1024);
  float* qpl = allocF((size_t)Nn*192);  float* qpg = allocF((size_t)Nn*192);
  float* kpl = allocF((size_t)Nn*192);  float* kpg = allocF((size_t)Nn*192);
  float* vpl = allocF((size_t)Nn*288);  float* vpg = allocF((size_t)Nn*288);
  float* ebias = allocF((size_t)E*8);
  float* logit = allocF((size_t)E*8);
  float* araw  = allocF((size_t)E*8);
  unsigned* segmax = allocU((size_t)Nn*8);
  float* segsum = allocF((size_t)Nn*8);
  float* o_  = allocF((size_t)Nn*1024);
  float* oe_ = allocF((size_t)Nn*1024);
  float* op_ = allocF((size_t)Nn*288);
  float* opl = allocF((size_t)Nn*288);
  __bf16* catb = allocB((size_t)Nn*2432);
  float* supd = allocF((size_t)Nn*256);
  float* vupd = allocF((size_t)Nn*48);
  __bf16* cat272 = allocB((size_t)Nn*288);   // padded 272 -> 288
  float* gate = allocF((size_t)Nn*16);
  __bf16* nsb = allocB((size_t)Nn*128);
  __bf16* xeb = allocB((size_t)E*384);
  __bf16* heb1 = allocB((size_t)E*128);
  float* h2e = allocF((size_t)E*128);
  __bf16* hnb1 = allocB((size_t)Nn*256);
  __bf16* hnb2 = allocB((size_t)Nn*256);
  float* h2n = allocF((size_t)Nn*256);
  __bf16* nodein = allocB((size_t)Nn*352);   // padded 321 -> 352
  float* psi = allocF((size_t)Nn*2);
  float* upd6 = allocF((size_t)Nn*6);

  auto gemm = [&](const __bf16* A, const __bf16* B, float* Df, __bf16* Dbf, int M, int Nc, int K, int act) {
    dim3 g((unsigned)(M / 64), (unsigned)((Nc + 63) / 64));
    k_gemm<<<g, 256, 0, stream>>>(A, B, Df, Dbf, M, Nc, K, act);
  };
  auto fillf = [&](float* p, float v, long long n) { k_fillf<<<gb(n), 256, 0, stream>>>(p, v, n); };
  auto fillu = [&](unsigned* p, unsigned v, long long n) { k_fillu<<<gb(n), 256, 0, stream>>>(p, v, n); };

  // ---- init frames + features ----
  k_center<<<1, 256, 0, stream>>>(rig, center, Nn);
  k_initframes<<<gb(Nn), 256, 0, stream>>>(rig, center, xm, q0, tr, trc, keep, Nn);
  k_spfeat<<<gb((long long)E  * 128), 256, 0, stream>>>(trc, ei,  residx, E,  ef);
  k_spfeat<<<gb((long long)E2 * 128), 256, 0, stream>>>(trc, sei, residx, E2, sef);

  auto embed_edges = [&](float* eff, __bf16* efbb, int Ec) {
    k_cvt<<<gb((long long)Ec * 128), 256, 0, stream>>>(eff, efbb, (long long)Ec * 128);
    gemm(efbb, eew1, nullptr, heb1, Ec, 128, 128, 1);
    gemm(heb1, eew2, nullptr, xeb,  Ec, 128, 128, 1);   // xeb reused as bf16 scratch
    gemm(xeb,  eew3, h2e, nullptr,  Ec, 128, 128, 0);
    k_ln<<<dim3((unsigned)Ec), 256, 0, stream>>>(h2e, nullptr, nullptr, F(1), F(0), eff, efbb, 128);
  };
  embed_edges(ef,  efb,  E);
  embed_edges(sef, sefb, E2);

  k_nodeinput<<<gb((long long)Nn * 352), 256, 0, stream>>>(residx, tarr, nm, nodein, Nn);
  gemm(nodein, enw1, nullptr, hnb1, Nn, 256, 352, 1);   // K = 352 (padded 321)
  gemm(hnb1,   enw2, nullptr, hnb2, Nn, 256, 256, 1);
  gemm(hnb2,   enw3, h2n, nullptr,  Nn, 256, 256, 0);
  k_ln<<<dim3((unsigned)Nn), 256, 0, stream>>>(h2n, nullptr, nullptr, F(6), F(5), s, sbf, 256);
  k_maskrow<<<gb((long long)Nn * 256), 256, 0, stream>>>(s, sbf, keep, 256, (long long)Nn * 256);

  float* nv = nvA; float* nv2 = nvB;

  auto run_attn = [&](const AttnW& w, const float* eff, const __bf16* efbb,
                      const int* eidx, int Ec, bool use_nv) {
    gemm(sbf, w.wq,  nullptr, qbf, Nn, 1024, 256, 0);
    gemm(sbf, w.wk,  nullptr, kbf, Nn, 1024, 256, 0);
    gemm(sbf, w.wv,  vf, nullptr,  Nn, 1024, 256, 0);
    gemm(sbf, w.wqp, qpl, nullptr, Nn, 192, 256, 0);
    gemm(sbf, w.wkp, kpl, nullptr, Nn, 192, 256, 0);
    gemm(sbf, w.wvp, vpl, nullptr, Nn, 288, 256, 0);
    if (use_nv) k_addvec<<<gb((long long)Nn * 288), 256, 0, stream>>>(vpl, nv, w.wvvec, Nn);
    k_toglobal<<<gb((long long)Nn * 64), 256, 0, stream>>>(q0, tr, qpl, qpg, Nn, 64);
    k_toglobal<<<gb((long long)Nn * 64), 256, 0, stream>>>(q0, tr, kpl, kpg, Nn, 64);
    k_toglobal<<<gb((long long)Nn * 96), 256, 0, stream>>>(q0, tr, vpl, vpg, Nn, 96);
    gemm(efbb, w.web, ebias, nullptr, Ec, 8, 128, 0);
    fillu(segmax, 0u, (long long)Nn * 8);
    fillf(segsum, 0.f, (long long)Nn * 8);
    k_logits<<<dim3((unsigned)Ec), 256, 0, stream>>>(qbf, kbf, qpg, kpg, ebias, w.gamma, eidx, Ec, logit, segmax);
    k_expsum<<<gb((long long)Ec * 8), 256, 0, stream>>>(logit, segmax, eidx, Ec, araw, segsum);
    fillf(o_,  0.f, (long long)Nn * 1024);
    fillf(oe_, 0.f, (long long)Nn * 1024);
    fillf(op_, 0.f, (long long)Nn * 288);
    k_scatter<<<dim3((unsigned)Ec), 256, 0, stream>>>(araw, segsum, eidx, Ec, vf, eff, vpg, o_, oe_, op_);
    k_cat<<<dim3((unsigned)Nn), 256, 0, stream>>>(o_, oe_, op_, q0, tr, opl, catb, Nn);
    gemm(catb, w.wout, supd, nullptr, Nn, 256, 2432, 0);
    k_vupd<<<gb((long long)Nn * 48), 256, 0, stream>>>(opl, w.wvout, vupd, Nn);
  };

  auto run_local = [&](const LuW& w) {
    k_cat272<<<gb((long long)Nn * 288), 256, 0, stream>>>(s, nv, cat272, Nn);
    gemm(cat272, w.w1, nullptr, hnb1, Nn, 128, 288, 1);   // K = 288 (padded 272)
    gemm(hnb1, w.ws, supd, nullptr, Nn, 256, 128, 0);
    gemm(hnb1, w.wg, gate, nullptr, Nn, 16, 128, 0);
    k_gatevupd<<<gb((long long)Nn * 48), 256, 0, stream>>>(gate, nv, w.wv, vupd, Nn);
  };

  auto run_et = [&](const EtW& w, float* eff, __bf16* efbb, const int* eidx, int Ec) {
    gemm(sbf, w.wn, nullptr, nsb, Nn, 128, 256, 0);
    k_edgecat<<<gb((long long)Ec * 384), 256, 0, stream>>>(nsb, efbb, eidx, Ec, xeb);
    gemm(xeb, w.w1, nullptr, heb1, Ec, 128, 384, 1);
    gemm(heb1, w.w2, h2e, nullptr, Ec, 128, 128, 0);
    k_ln<<<dim3((unsigned)Ec), 256, 0, stream>>>(h2e, eff, nullptr, w.g, w.b, eff, efbb, 128);
  };

  for (int l = 0; l < 4; ++l) {
    int lb = 10 + l * 56;
    run_attn(aseq[l], sef, sefb, sei, E2, l > 0);
    k_ln<<<dim3((unsigned)Nn), 256, 0, stream>>>(supd, s, keep, F(lb+33), F(lb+34), s, sbf, 256);
    k_vln<<<dim3((unsigned)Nn), 64, 0, stream>>>(l > 0 ? nv : nullptr, vupd, keep, F(lb+53), nv);
    run_attn(aspw[l], ef, efb, ei, E, true);
    k_ln<<<dim3((unsigned)Nn), 256, 0, stream>>>(supd, s, keep, F(lb+35), F(lb+36), s, sbf, 256);
    k_vln<<<dim3((unsigned)Nn), 64, 0, stream>>>(nv, vupd, keep, F(lb+54), nv);
    run_local(lcuw[l]);
    k_addscaled<<<gb((long long)Nn * 256), 256, 0, stream>>>(s, supd, keep, 256, (long long)Nn * 256);
    k_addscaled<<<gb((long long)Nn * 48), 256, 0, stream>>>(nv, vupd, keep, 48, (long long)Nn * 48);
    run_local(lruw[l]);
    k_ln<<<dim3((unsigned)Nn), 256, 0, stream>>>(supd, s, keep, F(lb+37), F(lb+38), s, sbf, 256);
    k_vln<<<dim3((unsigned)Nn), 64, 0, stream>>>(nv, vupd, keep, F(lb+55), nv);
    gemm(sbf, ntw1[l], nullptr, hnb1, Nn, 256, 256, 1);
    gemm(hnb1, ntw2[l], h2n, nullptr, Nn, 256, 256, 0);
    k_ln<<<dim3((unsigned)Nn), 256, 0, stream>>>(h2n, s, nullptr, F(lb+44), F(lb+43), s, sbf, 256);
    k_ntv<<<gb((long long)Nn * 48), 256, 0, stream>>>(nv, F(lb+47), nv2, Nn);
    { float* tp = nv; nv = nv2; nv2 = tp; }
    k_maskrow<<<gb((long long)Nn * 256), 256, 0, stream>>>(s, sbf, keep, 256, (long long)Nn * 256);
    k_maskrow<<<gb((long long)Nn * 48), 256, 0, stream>>>(nv, (__bf16*)nullptr, keep, 48, (long long)Nn * 48);
    k_upd6<<<gb((long long)Nn * 6), 256, 0, stream>>>(s, nv, nm, F(lb+22), F(lb+23), upd6, Nn);
    k_frameupd<<<gb(Nn), 256, 0, stream>>>(q0, tr, upd6, Nn);
    run_et(etw[l],  ef,  efb,  ei,  E);
    run_et(setw[l], sef, sefb, sei, E2);
  }

  gemm(sbf, torw1, nullptr, hnb1, Nn, 128, 256, 1);
  gemm(hnb1, torw2, psi, nullptr, Nn, 2, 128, 0);
  k_out<<<dim3((unsigned)Nn), 256, 0, stream>>>(s, q0, tr, center, psi, outp, Nn);
}
